// EpisB_24111946399760
// MI455X (gfx1250) — matrix operand order repaired and checked by reading, not compiled, not run on hardware
//
#include <hip/hip_runtime.h>

typedef __bf16 bf16_t;
typedef __attribute__((ext_vector_type(16))) __bf16 v16bf;
typedef __attribute__((ext_vector_type(8)))  __bf16 v8bf;
typedef __attribute__((ext_vector_type(8)))  float  v8f;

#define NN 8192
#define HH 4
#define TT 64

// ---------------------------------------------------------------------------
// Output #3: out3[j,i] = sigmoid(Amat[j,i])/10 + (i==j)   (transposes cancel)
// Also cache M[j,i] = sigmoid(Amat[j,i])/10 as bf16 (128 MB -> L2 resident).
// ---------------------------------------------------------------------------
__global__ void prep_matrix(const float* __restrict__ Amat,
                            float* __restrict__ out3,
                            bf16_t* __restrict__ Mbf) {
    long p4 = ((long)blockIdx.x * blockDim.x + threadIdx.x) * 4;
    const float4 a4 = *(const float4*)(Amat + p4);
    float vals[4] = {a4.x, a4.y, a4.z, a4.w};
    float o[4];
#pragma unroll
    for (int k = 0; k < 4; ++k) {
        long p = p4 + k;
        int j = (int)(p >> 13);
        int i = (int)(p & (NN - 1));
        float s = 0.1f / (1.0f + __expf(-vals[k]));
        o[k] = s + ((i == j) ? 1.0f : 0.0f);
        Mbf[p] = (bf16_t)s;
    }
    *(float4*)(out3 + p4) = make_float4(o[0], o[1], o[2], o[3]);
}

// ---------------------------------------------------------------------------
// Output #2: signal = relu(x); init S = 1 - signal[:,:,0], I = signal[:,:,0]
// ---------------------------------------------------------------------------
__global__ void prep_signal(const float* __restrict__ x,
                            float* __restrict__ sig,
                            float* __restrict__ Sws,
                            float* __restrict__ Iws) {
    int p = blockIdx.x * blockDim.x + threadIdx.x;   // 0 .. N*H*T-1
    float v = x[p];
    float r = v > 0.0f ? v : 0.0f;
    sig[p] = r;
    if ((p & (TT - 1)) == 0) {                        // t == 0 slice
        int idx = p >> 6;                             // (j*H + h)
        Iws[idx] = r;
        Sws[idx] = 1.0f - r;
    }
}

// ---------------------------------------------------------------------------
// Pack alpha = 1 - exp(-R0dTaus * I) into this code's contiguous-k WMMA B-fragment order:
// Bpack[(c*32 + lane)*16 + t] = alpha[32c + 16*(lane>>4) + t, lane&15]
// (columns n >= 4 are zero). One thread per (chunk, lane-slot) = 8192 threads.
// ---------------------------------------------------------------------------
__global__ void pack_alpha(const float* __restrict__ Iws,
                           const float* __restrict__ R0dTaus,
                           bf16_t* __restrict__ Bpack) {
    int tid = blockIdx.x * blockDim.x + threadIdx.x;  // 0..8191
    int c     = tid >> 5;
    int l     = tid & 31;
    int n     = l & 15;
    int khalf = l >> 4;
    bf16_t* dst = Bpack + (size_t)tid * 16;
    if (n < HH) {
        int ibase = c * 32 + khalf * 16;
#pragma unroll
        for (int t = 0; t < 16; ++t) {
            int idx = (ibase + t) * HH + n;
            float a = 1.0f - __expf(-R0dTaus[idx] * Iws[idx]);
            dst[t] = (bf16_t)a;
        }
    } else {
#pragma unroll
        for (int t = 0; t < 16; ++t) dst[t] = (bf16_t)0.0f;
    }
}

// ---------------------------------------------------------------------------
// One scan step: Alpha = M @ alpha + alpha (fp32 identity term in epilogue),
// dS = Alpha*S; S -= dS; I += dS - I/taus; pred[:, :, t] = dS.
// One wave per 16-row block of j; K-reduction 8192 via 256 WMMAs.
// ---------------------------------------------------------------------------
__global__ void step_matmul(const bf16_t* __restrict__ Mbf,
                            const bf16_t* __restrict__ Bpack,
                            const float* __restrict__ taus,
                            const float* __restrict__ R0dTaus,
                            float* __restrict__ Sws,
                            float* __restrict__ Iws,
                            float* __restrict__ pred,
                            int t) {
    const int lane = threadIdx.x & 31;
    const int wave = blockIdx.x * (blockDim.x >> 5) + (threadIdx.x >> 5);
    const int jb   = wave;                       // 0..511
    const int row  = jb * 16 + (lane & 15);      // A-matrix row for this lane
    const int colbase = (lane >> 4) ? 16 : 0;    // K sub-offset per lane half
    const bf16_t* Arow = Mbf + (size_t)row * NN;

    v8f acc = {};
#pragma unroll 4
    for (int c = 0; c < 256; ++c) {
        const int i0 = c * 32;
        // A fragment: lanes 0-15 hold K {0..15}; lanes 16-31 {16..31}
        v8bf alo = *(const v8bf*)(Arow + i0 + colbase);
        v8bf ahi = *(const v8bf*)(Arow + i0 + colbase + 8);
        v16bf a;
#pragma unroll
        for (int e = 0; e < 8; ++e) { a[e] = alo[e]; a[8 + e] = ahi[e]; }
        // B fragment: pre-swizzled, 32B contiguous per lane
        v16bf b = *(const v16bf*)(Bpack + ((size_t)c * 32 + lane) * 16);
        acc = __builtin_amdgcn_wmma_f32_16x16x32_bf16(
                  false, a, false, b, (short)0, acc, false, false);
    }

    // Epilogue: C layout -> VGPR r, lanes 0-15: M=r; lanes 16-31: M=8+r; N=lane&15
    const int n = lane & 15;
    if (n < HH) {
        const int mbase = (lane >> 4) * 8;
#pragma unroll
        for (int r = 0; r < 8; ++r) {
            int j = jb * 16 + mbase + r;
            int idx = j * HH + n;
            float Iv = Iws[idx];
            float Sv = Sws[idx];
            float aself = 1.0f - __expf(-R0dTaus[idx] * Iv);  // identity term
            float Alpha = acc[r] + aself;
            float dS = Alpha * Sv;
            Sws[idx] = Sv - dS;
            Iws[idx] = Iv - Iv / taus[idx] + dS;
            pred[(size_t)idx * TT + t] = dS;
        }
    }
}

extern "C" void kernel_launch(void* const* d_in, const int* in_sizes, int n_in,
                              void* d_out, int out_size, void* d_ws, size_t ws_size,
                              hipStream_t stream) {
    const float* x        = (const float*)d_in[0];
    const float* Amat     = (const float*)d_in[1];
    const float* taus     = (const float*)d_in[2];
    const float* R0dTaus  = (const float*)d_in[3];

    float* out  = (float*)d_out;
    float* pred = out;                                   // (N,H,T)
    float* sig  = out + (size_t)NN * HH * TT;            // (N,H,T)
    float* out3 = out + (size_t)2 * NN * HH * TT;        // (N,N)

    char* ws = (char*)d_ws;
    bf16_t* Mbf   = (bf16_t*)ws;                         // N*N*2   = 128 MB
    float*  Sws   = (float*)(ws + (size_t)NN * NN * 2);  // N*H*4
    float*  Iws   = Sws + (size_t)NN * HH;               // N*H*4
    bf16_t* Bpack = (bf16_t*)(Iws + (size_t)NN * HH);    // 8192*16*2 = 256 KB

    // One pass over Amat: fp32 output #3 + bf16 matmul operand cache.
    prep_matrix<<<(NN * (size_t)NN) / 4 / 256, 256, 0, stream>>>(Amat, out3, Mbf);
    // relu(x) output + S/I state init.
    prep_signal<<<(NN * HH * TT) / 256, 256, 0, stream>>>(x, sig, Sws, Iws);

    // Sequential scan: global reduction each step -> per-step launches.
    for (int t = 0; t < TT; ++t) {
        pack_alpha<<<32, 256, 0, stream>>>(Iws, R0dTaus, Bpack);
        step_matmul<<<128, 128, 0, stream>>>(Mbf, Bpack, taus, R0dTaus,
                                             Sws, Iws, pred, t);
    }
}
